// VoxelHashTableDynamicMulti_738734375107
// MI455X (gfx1250) — compile-verified, matrix operand order checked by reading, not run on hardware
//
#include <hip/hip_runtime.h>
#include <stdint.h>

// ---------------------------------------------------------------------------
// Types for CDNA5 WMMA (gfx1250, wave32)
// ---------------------------------------------------------------------------
typedef __bf16 bf16;
typedef __attribute__((ext_vector_type(16))) __bf16 v16bf;
typedef __attribute__((ext_vector_type(8)))  float  v8f;

#define HTS_MASK ((1u << 20) - 1u)
#define DM   120
#define MODT 201

// ---------------------------------------------------------------------------
// A fragment per CDNA5 ISA 7.12.2 (16-bit A 16x32 layout).
// Element e of the per-lane v16bf holds K index:
//   k = (e & 7) + 8*(lane>=16) + 16*(e>>3)
// All indexing in 32-bit ints off the original LDS array pointer so the
// compiler's address-space inference keeps these as plain ds_load_b128.
// ---------------------------------------------------------------------------
__device__ inline v16bf load_frag_a(const bf16* A, int base, int lda, int lane) {
  int m  = lane & 15;
  int hi = (lane >> 4) & 1;
  v16bf a = {};
#pragma unroll
  for (int e = 0; e < 16; ++e) {
    int k = (e & 7) + 8 * hi + 16 * (e >> 3);
    a[e] = A[base + m * lda + k];
  }
  return a;
}

// B operand pre-packed fragment-major: per (ktile, ntile) 512 bf16, lane's
// 16 elements contiguous (32 bytes). Uniform tile index + lane -> saddr
// global_load_b128 pair, no pack VALU.
__device__ inline v16bf load_frag_bp(const bf16* __restrict__ Bpack,
                                     int nTilesN, int kt, int nt, int lane) {
  const v16bf* Bv = (const v16bf*)Bpack;
  return Bv[(kt * nTilesN + nt) * 32 + lane];
}

__device__ inline v8f wmma_acc(v16bf a, v16bf b, v8f c) {
  return __builtin_amdgcn_wmma_f32_16x16x32_bf16(false, a, false, b,
                                                 (short)0, c, false, false);
}

// C/D layout: vgpr r -> (m = r + 8*(lane>=16), n = lane&15)
__device__ inline void store_tile(v8f c, const float* __restrict__ bias,
                                  int nbase, bf16* dst, int dbase, int ldd,
                                  int ncols, bool relu, int lane) {
  int n   = lane & 15;
  int col = nbase + n;
  int mo  = ((lane >> 4) & 1) * 8;
  if (col < ncols) {
    float bv = bias[col];
#pragma unroll
    for (int r = 0; r < 8; ++r) {
      float x = c[r] + bv;
      if (relu && x < 0.0f) x = 0.0f;
      dst[dbase + (mo + r) * ldd + col] = (bf16)x;
    }
  }
}

// D[16*Mt x Np] = A[16*Mt x K] * B[K x Np] + bias  (bf16 LDS dest).
// A fragments hoisted into registers once per 16-row block (K/32 frags),
// so the n-tile loop does only B loads + chained WMMA.
template <int K>
__device__ inline void gemm_bf16(const bf16* A, int aBase, int lda, int Mt,
                                 const bf16* __restrict__ Bpack, int Np,
                                 const float* __restrict__ bias,
                                 bf16* D, int dBase, int ldd,
                                 int ncols, bool relu, int lane) {
  constexpr int KT = K / 32;
  const int nTiles = Np / 16;
  for (int mt = 0; mt < Mt; ++mt) {
    int ab = aBase + mt * 16 * lda;
    int db = dBase + mt * 16 * ldd;
    v16bf afr[KT];
#pragma unroll
    for (int kt = 0; kt < KT; ++kt) afr[kt] = load_frag_a(A, ab + kt * 32, lda, lane);
    for (int nt = 0; nt < nTiles; ++nt) {
      v8f c = {};
#pragma unroll
      for (int kt = 0; kt < KT; ++kt)
        c = wmma_acc(afr[kt], load_frag_bp(Bpack, nTiles, kt, nt, lane), c);
      store_tile(c, bias, nt * 16, D, db, ldd, ncols, relu, lane);
    }
  }
}

// ---------------------------------------------------------------------------
// Parameter block
// ---------------------------------------------------------------------------
struct FuseW {
  const bf16* Wq; const bf16* Wk; const bf16* Wv; const bf16* Wo;
  const float* bq; const float* bk; const float* bv; const float* bo;
};

struct Params {
  const float* qpts;  const float* qpose; const float* qstate;
  const float* statf; const float* dynT;  const float* dynP; const float* dynS;
  const float* temb;
  const bf16* poseW1;  const float* pose_b1;
  const bf16* poseW2;  const float* pose_b2;
  const bf16* stateW1; const float* state_b1;
  const bf16* stateW2; const float* state_b2;
  FuseW fT, fP, fS, fA;
  const int* times; const int* buf;
  float* out;
  int Mq;
};

// ---------------------------------------------------------------------------
// Gather 16 feature rows (f32 global -> bf16 LDS, dst stride 128).
// Rows are 480 B (16B aligned) -> float4 loads, 8-byte bf16x4 LDS stores.
// ---------------------------------------------------------------------------
__device__ inline void gather_rows(bf16* dst, int dBase,
                                   const float* __restrict__ src,
                                   const int* rows, int lane) {
  for (int i = lane; i < 16 * 30; i += 32) {
    int q = i / 30, c4 = i % 30;
    const float4 v = ((const float4*)(src + (size_t)rows[q] * DM))[c4];
    int d = dBase + q * 128 + c4 * 4;
    dst[d + 0] = (bf16)v.x; dst[d + 1] = (bf16)v.y;
    dst[d + 2] = (bf16)v.z; dst[d + 3] = (bf16)v.w;
  }
}

// ---------------------------------------------------------------------------
// Multi-head attention fusion over S tokens (S = 2 or 4), mean-pooled.
// tok: (S*16) x 128 bf16 LDS (row = s*16 + q), Qb/Kb: 64x128 bf16 scratch.
// condDst != nullptr -> write 16x120 bf16 cond rows at condBase; else final out.
// ---------------------------------------------------------------------------
__device__ void fuse_mha(const bf16* tok, int S, const FuseW w,
                         bf16* Qb, bf16* Kb, bf16* condDst, int condBase,
                         float* outG, const int* validq, int qbase, int Mq,
                         int lane) {
  __syncthreads();
  // Q,K projections sharing hoisted A fragments; each n-tile loop keeps a
  // single uniform B base (saddr global_load form).
  for (int mt = 0; mt < S; ++mt) {
    int ab = mt * 16 * 128;
    v16bf afr[4];
#pragma unroll
    for (int kt = 0; kt < 4; ++kt) afr[kt] = load_frag_a(tok, ab + kt * 32, 128, lane);
    for (int nt = 0; nt < 8; ++nt) {
      v8f c = {};
#pragma unroll
      for (int kt = 0; kt < 4; ++kt)
        c = wmma_acc(afr[kt], load_frag_bp(w.Wq, 8, kt, nt, lane), c);
      store_tile(c, w.bq, nt * 16, Qb, ab, 128, DM, false, lane);
    }
    for (int nt = 0; nt < 8; ++nt) {
      v8f c = {};
#pragma unroll
      for (int kt = 0; kt < 4; ++kt)
        c = wmma_acc(afr[kt], load_frag_bp(w.Wk, 8, kt, nt, lane), c);
      store_tile(c, w.bk, nt * 16, Kb, ab, 128, DM, false, lane);
    }
  }
  __syncthreads();

  // Per-lane attention: 16 queries x 8 heads = 128 pairs, 4 per lane.
  const float sc = 0.25819888974716113f;  // 15^-0.5
  float att[4][16];
#pragma unroll
  for (int it = 0; it < 4; ++it) {
    int p = lane + 32 * it;
    int q = p >> 3, h = p & 7, c0 = h * 15;
    float s[16];
    for (int si = 0; si < S; ++si)
      for (int ti = 0; ti < S; ++ti) {
        float acc = 0.0f;
#pragma unroll
        for (int d = 0; d < 15; ++d)
          acc += (float)Qb[(si * 16 + q) * 128 + c0 + d] *
                 (float)Kb[(ti * 16 + q) * 128 + c0 + d];
        s[si * 4 + ti] = acc * sc;
      }
    for (int si = 0; si < S; ++si) {
      float mx = -1e30f;
      for (int ti = 0; ti < S; ++ti) mx = fmaxf(mx, s[si * 4 + ti]);
      float sum = 0.0f;
      for (int ti = 0; ti < S; ++ti) {
        float e = __expf(s[si * 4 + ti] - mx);
        s[si * 4 + ti] = e;
        sum += e;
      }
      float inv = 1.0f / sum;
      for (int ti = 0; ti < S; ++ti) att[it][si * 4 + ti] = s[si * 4 + ti] * inv;
    }
  }
  __syncthreads();

  // V projection overwrites K buffer
  gemm_bf16<128>(tok, 0, 128, S, w.Wv, 128, w.bv, Kb, 0, 128, DM, false, lane);
  __syncthreads();

  // O = att @ V, written over Q buffer (cols 120..127 stay benign: Wo pad rows 0)
#pragma unroll
  for (int it = 0; it < 4; ++it) {
    int p = lane + 32 * it;
    int q = p >> 3, h = p & 7, c0 = h * 15;
    for (int si = 0; si < S; ++si)
#pragma unroll
      for (int d = 0; d < 15; ++d) {
        float o = 0.0f;
        for (int ti = 0; ti < S; ++ti)
          o += att[it][si * 4 + ti] * (float)Kb[(ti * 16 + q) * 128 + c0 + d];
        Qb[(si * 16 + q) * 128 + c0 + d] = (bf16)o;
      }
  }
  __syncthreads();

  // Output projection + mean over tokens: chain WMMA accumulators over si.
  // Two halves of 4 n-tiles each to keep live accumulators at 32 VGPRs.
  float invS = 1.0f / (float)S;
  int n = lane & 15, mo = ((lane >> 4) & 1) * 8;
  for (int half = 0; half < 2; ++half) {
    v8f acc[4] = {};
    for (int si = 0; si < S; ++si) {
      int ab = si * 16 * 128;
      v16bf afr[4];
#pragma unroll
      for (int kt = 0; kt < 4; ++kt) afr[kt] = load_frag_a(Qb, ab + kt * 32, 128, lane);
#pragma unroll
      for (int nh = 0; nh < 4; ++nh)
#pragma unroll
        for (int kt = 0; kt < 4; ++kt)
          acc[nh] = wmma_acc(afr[kt],
                             load_frag_bp(w.Wo, 8, kt, half * 4 + nh, lane),
                             acc[nh]);
    }
#pragma unroll
    for (int nh = 0; nh < 4; ++nh) {
      int col = (half * 4 + nh) * 16 + n;
      if (col < DM) {
        float bv = w.bo[col];
#pragma unroll
        for (int r = 0; r < 8; ++r) {
          int q = mo + r;
          float val = acc[nh][r] * invS + bv;
          if (condDst) {
            condDst[condBase + q * 128 + col] = (bf16)val;
          } else {
            int qg = qbase + q;
            if (qg < Mq) outG[(size_t)qg * DM + col] = validq[q] ? val : 0.0f;
          }
        }
      }
    }
  }
  __syncthreads();
}

// ---------------------------------------------------------------------------
// Weight pack: f32 (K x N) -> bf16 WMMA-B-fragment-major, zero padded to Kp x Np.
// Per (ktile, ntile) 512-elem tile: offset = lane*16 + e, with
//   lane = (n%16) + 16*((k>>3)&1),  e = (k&7) + 8*((k>>4)&1)
// which inverts the fragment mapping k = (e&7) + 8*hi + 16*(e>>3).
// ---------------------------------------------------------------------------
__global__ void pack_b_kernel(const float* __restrict__ src,
                              bf16* __restrict__ dst,
                              int K, int N, int Kp, int Np) {
  int i = blockIdx.x * blockDim.x + threadIdx.x;
  if (i >= Kp * Np) return;
  int k = i / Np, n = i % Np;
  float val = (k < K && n < N) ? src[(size_t)k * N + n] : 0.0f;
  int kt = k >> 5, kk = k & 31;
  int ntile = n >> 4, nn = n & 15;
  int hi = (kk >> 3) & 1;
  int e  = (kk & 7) + 8 * ((kk >> 4) & 1);
  int lane = nn + 16 * hi;
  size_t off = ((size_t)(kt * (Np >> 4) + ntile) * 512) + lane * 16 + e;
  dst[off] = (bf16)val;
}

// ---------------------------------------------------------------------------
// Main fused kernel: one wave32 per 16-query tile. Cap at 256 VGPRs
// (waves_per_eu(4)) to avoid VGPR-MSB switching and keep occupancy.
// ---------------------------------------------------------------------------
__global__ __launch_bounds__(32)
__attribute__((amdgpu_waves_per_eu(4)))
void voxel_fuse_kernel(Params P) {
  __shared__ bf16 tokT[64 * 128];   // final 4-token matrix (static|condT|condP|condS)
  __shared__ bf16 tokS[32 * 128];   // 2-token matrix for the small fusions
  __shared__ bf16 Qb[64 * 128];     // Q / MLP-hidden / O scratch
  __shared__ bf16 Kb[64 * 128];     // K / V scratch
  __shared__ bf16 xbuf[16 * 64];    // MLP input (K padded to 32/64)
  __shared__ int  vq[16], tq[16], validq[16];

  const int lane  = threadIdx.x;
  const int qbase = blockIdx.x * 16;

  // Zero only what WMMA needs finite/zero: pad columns 120..127 of the four
  // 128-stride buffers (224 rows total) + the whole MLP input staging buffer.
  for (int r = lane; r < 224; r += 32) {
    bf16* base; int rb;
    if      (r < 64)  { base = tokT; rb = r * 128; }
    else if (r < 96)  { base = tokS; rb = (r - 64) * 128; }
    else if (r < 160) { base = Qb;   rb = (r - 96) * 128; }
    else              { base = Kb;   rb = (r - 160) * 128; }
#pragma unroll
    for (int c = 0; c < 8; ++c) base[rb + 120 + c] = (bf16)0.0f;
  }
  for (int i = lane; i < 16 * 64; i += 32) xbuf[i] = (bf16)0.0f;

  // ---- hash lookup (lanes 0..15, one query each) ----
  if (lane < 16) {
    int qg = qbase + lane;
    int vi = -1, tt = 0, ok = 0;
    if (qg < P.Mq) {
      float px = P.qpts[3 * qg + 0];
      float py = P.qpts[3 * qg + 1];
      float pz = P.qpts[3 * qg + 2];
      int gx = (int)floorf(px / 0.1f);
      int gy = (int)floorf(py / 0.1f);
      int gz = (int)floorf(pz / 0.1f);
      unsigned h = (unsigned)gx * 73856093u + (unsigned)gy * 19349669u +
                   (unsigned)gz * 83492791u;
      vi = P.buf[h & HTS_MASK];
      tt = P.times[qg] % MODT;
      ok = (vi >= 0);
    }
    validq[lane] = ok;
    vq[lane]     = (vi >= 0) ? vi : 0;
    tq[lane]     = tt;
  }
  __syncthreads();

  int nq = P.Mq - qbase; if (nq > 16) nq = 16;

  // ================= fusion_time: [dyn_time[v], time_emb[t]] =================
  gather_rows(tokS, 0,        P.dynT, vq, lane);
  gather_rows(tokS, 16 * 128, P.temb, tq, lane);
  fuse_mha(tokS, 2, P.fT, Qb, Kb, tokT, 16 * 128, nullptr, validq, qbase, P.Mq, lane);

  // ================= pose MLP + fusion_pose =================
  for (int i = lane; i < nq * 12; i += 32) {
    int q = i / 12, c = i % 12;
    xbuf[q * 64 + c] = (bf16)P.qpose[(size_t)(qbase + q) * 12 + c];
  }
  __syncthreads();
  // hidden = relu(x @ W1 + b1) : 16x32 * 32x256, stored in Qb (stride 256)
  gemm_bf16<32>(xbuf, 0, 64, 1, P.poseW1, 256, P.pose_b1, Qb, 0, 256, 256, true, lane);
  __syncthreads();
  // feat = hidden @ W2 + b2 -> token 1 rows of tokS
  gemm_bf16<256>(Qb, 0, 256, 1, P.poseW2, 128, P.pose_b2, tokS, 16 * 128, 128, DM, false, lane);
  gather_rows(tokS, 0, P.dynP, vq, lane);
  fuse_mha(tokS, 2, P.fP, Qb, Kb, tokT, 32 * 128, nullptr, validq, qbase, P.Mq, lane);

  // ================= state MLP + fusion_state =================
  for (int i = lane; i < nq * 42; i += 32) {
    int q = i / 42, c = i % 42;
    xbuf[q * 64 + c] = (bf16)P.qstate[(size_t)(qbase + q) * 42 + c];
  }
  __syncthreads();
  gemm_bf16<64>(xbuf, 0, 64, 1, P.stateW1, 256, P.state_b1, Qb, 0, 256, 256, true, lane);
  __syncthreads();
  gemm_bf16<256>(Qb, 0, 256, 1, P.stateW2, 128, P.state_b2, tokS, 16 * 128, 128, DM, false, lane);
  gather_rows(tokS, 0, P.dynS, vq, lane);
  fuse_mha(tokS, 2, P.fS, Qb, Kb, tokT, 48 * 128, nullptr, validq, qbase, P.Mq, lane);

  // ================= final fusion over 4 tokens =================
  gather_rows(tokT, 0, P.statf, vq, lane);
  fuse_mha(tokT, 4, P.fA, Qb, Kb, nullptr, 0, P.out, validq, qbase, P.Mq, lane);
}

// ---------------------------------------------------------------------------
// Host-side launcher
// ---------------------------------------------------------------------------
extern "C" void kernel_launch(void* const* d_in, const int* in_sizes, int n_in,
                              void* d_out, int out_size, void* d_ws, size_t ws_size,
                              hipStream_t stream) {
  (void)n_in; (void)out_size; (void)ws_size;
  bf16* ws = (bf16*)d_ws;

  // bf16 packed-weight arena (element offsets); packed size == Kp*Np
  const size_t OFF_PW1 = 0;                      // 32x256
  const size_t OFF_PW2 = OFF_PW1 + 32 * 256;     // 256x128
  const size_t OFF_SW1 = OFF_PW2 + 256 * 128;    // 64x256
  const size_t OFF_SW2 = OFF_SW1 + 64 * 256;     // 256x128
  const size_t OFF_F   = OFF_SW2 + 256 * 128;    // 16 x (128x128)

  auto pack = [&](const void* src, size_t off, int K, int N, int Kp, int Np) {
    int tot = Kp * Np;
    pack_b_kernel<<<(tot + 255) / 256, 256, 0, stream>>>(
        (const float*)src, ws + off, K, N, Kp, Np);
  };

  // MLP weights (tuple order: W1, b1, W2, b2)
  pack(d_in[8],  OFF_PW1, 12, 256, 32, 256);
  pack(d_in[10], OFF_PW2, 256, 120, 256, 128);
  pack(d_in[12], OFF_SW1, 42, 256, 64, 256);
  pack(d_in[14], OFF_SW2, 256, 120, 256, 128);
  // fusion dicts (insertion order: Wq, Wk, Wv, Wo, bq, bk, bv, bo)
  for (int f = 0; f < 4; ++f)
    for (int j = 0; j < 4; ++j)
      pack(d_in[16 + 8 * f + j], OFF_F + (size_t)(f * 4 + j) * 128 * 128,
           120, 120, 128, 128);

  Params P;
  P.qpts   = (const float*)d_in[0];
  P.qpose  = (const float*)d_in[1];
  P.qstate = (const float*)d_in[2];
  P.statf  = (const float*)d_in[3];
  P.dynT   = (const float*)d_in[4];
  P.dynP   = (const float*)d_in[5];
  P.dynS   = (const float*)d_in[6];
  P.temb   = (const float*)d_in[7];
  P.poseW1 = ws + OFF_PW1;  P.pose_b1  = (const float*)d_in[9];
  P.poseW2 = ws + OFF_PW2;  P.pose_b2  = (const float*)d_in[11];
  P.stateW1 = ws + OFF_SW1; P.state_b1 = (const float*)d_in[13];
  P.stateW2 = ws + OFF_SW2; P.state_b2 = (const float*)d_in[15];
  FuseW* fw[4] = { &P.fT, &P.fP, &P.fS, &P.fA };
  for (int f = 0; f < 4; ++f) {
    const bf16* base = ws + OFF_F + (size_t)f * 4 * 128 * 128;
    fw[f]->Wq = base + 0 * 128 * 128;
    fw[f]->Wk = base + 1 * 128 * 128;
    fw[f]->Wv = base + 2 * 128 * 128;
    fw[f]->Wo = base + 3 * 128 * 128;
    fw[f]->bq = (const float*)d_in[16 + 8 * f + 4];
    fw[f]->bk = (const float*)d_in[16 + 8 * f + 5];
    fw[f]->bv = (const float*)d_in[16 + 8 * f + 6];
    fw[f]->bo = (const float*)d_in[16 + 8 * f + 7];
  }
  P.times = (const int*)d_in[48];
  P.buf   = (const int*)d_in[49];
  P.out   = (float*)d_out;
  P.Mq    = in_sizes[48];

  int nblocks = (P.Mq + 15) / 16;
  voxel_fuse_kernel<<<nblocks, 32, 0, stream>>>(P);
}
